// MultiHeadSelfAttention_5660766896579
// MI455X (gfx1250) — compile-verified, matrix operand order checked by reading, not run on hardware
//
#include <hip/hip_runtime.h>
#include <hip/hip_bf16.h>
#include <math.h>

#define D_MODEL   1024
#define NUM_HEADS 16
#define HEAD_DIM  64
#define BATCH     2
#define SEQ       2048
#define BS        (BATCH * SEQ)        // 4096 rows
#define BH        (BATCH * NUM_HEADS)  // 32 (b,h) pairs

typedef __attribute__((ext_vector_type(16))) __bf16 v16bf;
typedef __attribute__((ext_vector_type(8)))  __bf16 v8bf;
typedef __attribute__((ext_vector_type(8)))  float  v8f;

// ---------------------------------------------------------------- utilities

__device__ __forceinline__ __bf16 f32_to_bf16(float f) {
  union { float f; unsigned u; } c; c.f = f;
  unsigned r = c.u + 0x7FFFu + ((c.u >> 16) & 1u);   // round-to-nearest-even
  unsigned short h = (unsigned short)(r >> 16);
  __bf16 b; __builtin_memcpy(&b, &h, 2);
  return b;
}

// Loads a WMMA 16x32 bf16 A-fragment from a row-major [rows, ld] matrix.
// Also serves as the B-fragment loader when B is stored [N, K] row-major
// (i.e. we compute A @ B^T), since the lane/K packing is identical.
// Per CDNA5 ISA 7.12.2: lane = row (0..15), lane-half selects K-octet,
// slot s -> k = (s&7) + 8*(lane>>4) + 16*(s>>3).
__device__ __forceinline__ v16bf load_frag(const __bf16* __restrict__ base, int ld,
                                           int row_base, int col_base) {
  const unsigned lane = threadIdx.x & 31u;
  const int half = (int)(lane >> 4) & 1;
  const __bf16* rp = base + (size_t)(row_base + (int)(lane & 15u)) * (size_t)ld
                          + col_base + (half << 3);
  v8bf lo = *(const v8bf*)rp;          // k = 8*half + 0..7       (16B load)
  v8bf hi = *(const v8bf*)(rp + 16);   // k = 16 + 8*half + 0..7  (16B load)
  v16bf f;
#pragma unroll
  for (int i = 0; i < 8; ++i) { f[i] = lo[i]; f[8 + i] = hi[i]; }
  return f;
}

// ---------------------------------------------------------------- fp32 -> bf16

__global__ void cvt_bf16_kernel(const float* __restrict__ in,
                                __bf16* __restrict__ out, int n) {
  int i = blockIdx.x * blockDim.x + threadIdx.x;
  int stride = gridDim.x * blockDim.x;
  for (; i < n; i += stride) out[i] = f32_to_bf16(in[i]);
}

// ---------------------------------------------------------------- WMMA GEMM
// C[M,N] (fp32) = A[M,K] (bf16, row-major) @ B[N,K]^T (bf16, row-major [N,K]).
// One wave computes a 32(M) x 64(N) slab: 8 accumulators; each B fragment is
// reused by two A fragments (halves weight traffic vs 16-row tiles).

__global__ __launch_bounds__(256) void gemm_bf16_kernel(
    const __bf16* __restrict__ A, const __bf16* __restrict__ B,
    float* __restrict__ C, int M, int N, int K) {
  const int wave   = blockIdx.x * (blockDim.x >> 5) + (int)(threadIdx.x >> 5);
  const int nTiles = N >> 6;                       // 64-wide N slabs
  const int mT = wave / nTiles;                    // 32-row M slabs
  const int nT = wave % nTiles;
  if (mT * 32 >= M) return;

  v8f acc[2][4];
#pragma unroll
  for (int i = 0; i < 2; ++i)
#pragma unroll
    for (int j = 0; j < 4; ++j)
#pragma unroll
      for (int r = 0; r < 8; ++r) acc[i][j][r] = 0.0f;

  const unsigned lane = threadIdx.x & 31u;
  const int rowA = mT * 32 + (int)(lane & 15u);
  const int rowB = nT * 64 + (int)(lane & 15u);

  for (int k = 0; k < K; k += 32) {
    // speculative prefetch of next k-slab (global_prefetch_b8); wave-uniform cond
    if (k + 32 < K) {
      __builtin_prefetch(A + (size_t)rowA * K + k + 32, 0, 0);
      __builtin_prefetch(B + (size_t)rowB * K + k + 32, 0, 0);
    }
    v16bf a0 = load_frag(A, K, mT * 32,      k);
    v16bf a1 = load_frag(A, K, mT * 32 + 16, k);
#pragma unroll
    for (int j = 0; j < 4; ++j) {
      v16bf b = load_frag(B, K, nT * 64 + j * 16, k);
      acc[0][j] = __builtin_amdgcn_wmma_f32_16x16x32_bf16(
          false, a0, false, b, (short)0, acc[0][j], false, false);
      acc[1][j] = __builtin_amdgcn_wmma_f32_16x16x32_bf16(
          false, a1, false, b, (short)0, acc[1][j], false, false);
    }
  }

  // C/D layout: element (m, n): lane = n (+16 for m>=8), VGPR r -> m = r + 8*half
  const int half = (int)(lane >> 4) & 1;
  const int n0   = nT * 64 + (int)(lane & 15u);
#pragma unroll
  for (int i = 0; i < 2; ++i)
#pragma unroll
    for (int j = 0; j < 4; ++j)
#pragma unroll
      for (int r = 0; r < 8; ++r)
        C[(size_t)(mT * 32 + i * 16 + r + 8 * half) * (size_t)N + n0 + j * 16]
            = acc[i][j][r];
}

// ---------------------------------------------------------------- RoPE + pack
// src: fp32 [B*S, D_MODEL]; dst: bf16 [B*H, S, 64], RoPE applied per pair.

__global__ void rope_pack_kernel(const float* __restrict__ src,
                                 const int* __restrict__ pos,
                                 __bf16* __restrict__ dst) {
  int idx = blockIdx.x * blockDim.x + threadIdx.x;      // over B*S*H*32
  const int total = BATCH * SEQ * NUM_HEADS * (HEAD_DIM / 2);
  if (idx >= total) return;
  int i = idx & 31;  int t = idx >> 5;
  int h = t & 15;    t >>= 4;
  int s = t % SEQ;   int b = t / SEQ;

  size_t src_off = (size_t)(b * SEQ + s) * D_MODEL + h * HEAD_DIM + 2 * i;
  float e = src[src_off];
  float o = src[src_off + 1];
  float p = (float)pos[s];
  float inv = __powf(10000.0f, -(float)i / 32.0f);
  float sn, cs;
  __sincosf(p * inv, &sn, &cs);
  size_t dst_off = ((size_t)(b * NUM_HEADS + h) * SEQ + s) * HEAD_DIM + 2 * i;
  dst[dst_off]     = f32_to_bf16(e * cs - o * sn);
  dst[dst_off + 1] = f32_to_bf16(e * sn + o * cs);
}

// V: fp32 [B*S, D_MODEL] -> bf16 transposed per head [B*H, 64, S]
__global__ void vpack_kernel(const float* __restrict__ src,
                             __bf16* __restrict__ dst) {
  int idx = blockIdx.x * blockDim.x + threadIdx.x;      // over B*S*D
  if (idx >= BS * D_MODEL) return;
  int d = idx & 63;  int t = idx >> 6;
  int h = t & 15;    t >>= 4;
  int s = t % SEQ;   int b = t / SEQ;
  dst[((size_t)(b * NUM_HEADS + h) * HEAD_DIM + d) * SEQ + s] = f32_to_bf16(src[idx]);
}

// O: fp32 [B*H, S, 64] -> bf16 [B*S, D_MODEL]
__global__ void opack_kernel(const float* __restrict__ src,
                             __bf16* __restrict__ dst) {
  int idx = blockIdx.x * blockDim.x + threadIdx.x;      // dst-linear over B*S*D
  if (idx >= BS * D_MODEL) return;
  int d = idx & 63;  int t = idx >> 6;
  int h = t & 15;    t >>= 4;
  int s = t % SEQ;   int b = t / SEQ;
  dst[idx] = f32_to_bf16(
      src[((size_t)(b * NUM_HEADS + h) * SEQ + s) * HEAD_DIM + d]);
}

// ---------------------------------------------------------------- attention
// One wave handles 16 queries of one (b,h). Computes S^T = K·Q^T so the
// softmaxed P^T converts lane-locally into the B-fragment of O^T = V^T·P^T.

__global__ __launch_bounds__(256) void attn_kernel(
    const __bf16* __restrict__ Qh,   // [B*H, S, 64]
    const __bf16* __restrict__ Kh,   // [B*H, S, 64]
    const __bf16* __restrict__ Vt,   // [B*H, 64, S]
    float* __restrict__ Of) {        // [B*H, S, 64]
  const int wave   = blockIdx.x * (blockDim.x >> 5) + (int)(threadIdx.x >> 5);
  const int qTiles = SEQ / 16;
  const int qt = wave % qTiles;
  const int bh = wave / qTiles;
  if (bh >= BH) return;

  const unsigned lane = threadIdx.x & 31u;
  const int half = (int)(lane >> 4) & 1;
  const int nq   = (int)(lane & 15u);
  const int qBase = qt * 16;
  const int qIdx  = qBase + nq;               // this lane's query column

  const __bf16* Q = Qh + (size_t)bh * SEQ * HEAD_DIM;
  const __bf16* K = Kh + (size_t)bh * SEQ * HEAD_DIM;
  const __bf16* V = Vt + (size_t)bh * HEAD_DIM * SEQ;

  // Q fragments (B-operand of S^T), loaded once: d in [0,32) and [32,64)
  v16bf qf0 = load_frag(Q, HEAD_DIM, qBase, 0);
  v16bf qf1 = load_frag(Q, HEAD_DIM, qBase, 32);

  v8f o[4];
#pragma unroll
  for (int j = 0; j < 4; ++j)
#pragma unroll
    for (int r = 0; r < 8; ++r) o[j][r] = 0.0f;

  float mrun = -__builtin_inff();
  float lrun = 0.0f;
  const float scale = 0.125f;                 // 1/sqrt(64)
  const int nkt = (qBase + 16 + 31) / 32;     // causal: keys <= qBase+15

  for (int t = 0; t < nkt; ++t) {
    const int kB = t * 32;
    v8f s0, s1;
#pragma unroll
    for (int r = 0; r < 8; ++r) { s0[r] = 0.0f; s1[r] = 0.0f; }
    // S^T tiles: rows = keys kB..kB+15 (s0) and kB+16..kB+31 (s1)
    s0 = __builtin_amdgcn_wmma_f32_16x16x32_bf16(false, load_frag(K, HEAD_DIM, kB, 0),  false, qf0, (short)0, s0, false, false);
    s0 = __builtin_amdgcn_wmma_f32_16x16x32_bf16(false, load_frag(K, HEAD_DIM, kB, 32), false, qf1, (short)0, s0, false, false);
    s1 = __builtin_amdgcn_wmma_f32_16x16x32_bf16(false, load_frag(K, HEAD_DIM, kB + 16, 0),  false, qf0, (short)0, s1, false, false);
    s1 = __builtin_amdgcn_wmma_f32_16x16x32_bf16(false, load_frag(K, HEAD_DIM, kB + 16, 32), false, qf1, (short)0, s1, false, false);

    // causal mask + scale; per-column (query) max over the 32 keys
    float p0[8], p1[8];
    float mx = -__builtin_inff();
#pragma unroll
    for (int r = 0; r < 8; ++r) {
      int k0 = kB + r + 8 * half;
      int k1 = k0 + 16;
      p0[r] = (k0 <= qIdx) ? s0[r] * scale : -__builtin_inff();
      p1[r] = (k1 <= qIdx) ? s1[r] * scale : -__builtin_inff();
      mx = fmaxf(mx, fmaxf(p0[r], p1[r]));
    }
    mx = fmaxf(mx, __shfl_xor(mx, 16, 32));   // combine lane halves (keys split)

    float mnew  = fmaxf(mrun, mx);
    float alpha = __expf(mrun - mnew);
    float sum = 0.0f;
#pragma unroll
    for (int r = 0; r < 8; ++r) {
      p0[r] = __expf(p0[r] - mnew);
      p1[r] = __expf(p1[r] - mnew);
      sum += p0[r] + p1[r];
    }
    sum += __shfl_xor(sum, 16, 32);
    lrun = lrun * alpha + sum;
    mrun = mnew;

    // rescale O^T accumulators (per-query alpha is lane-local)
#pragma unroll
    for (int j = 0; j < 4; ++j)
#pragma unroll
      for (int r = 0; r < 8; ++r) o[j][r] *= alpha;

    // P^T B-fragment: lane-local repack of the two score tiles
    v16bf pf;
#pragma unroll
    for (int r = 0; r < 8; ++r) {
      pf[r]     = f32_to_bf16(p0[r]);   // keys kB +  8*half + r
      pf[8 + r] = f32_to_bf16(p1[r]);   // keys kB+16+8*half + r
    }

    // O^T[d, q] += V^T (A) x P^T (B); V^T is row-major [64, S]
#pragma unroll
    for (int j = 0; j < 4; ++j)
      o[j] = __builtin_amdgcn_wmma_f32_16x16x32_bf16(
          false, load_frag(V, SEQ, j * 16, kB), false, pf, (short)0, o[j], false, false);
  }

  const float inv = 1.0f / lrun;
  float* Op = Of + (size_t)bh * SEQ * HEAD_DIM;
#pragma unroll
  for (int j = 0; j < 4; ++j)
#pragma unroll
    for (int r = 0; r < 8; ++r)
      Op[(size_t)(qBase + nq) * HEAD_DIM + j * 16 + r + 8 * half] = o[j][r] * inv;
}

// ---------------------------------------------------------------- launcher

extern "C" void kernel_launch(void* const* d_in, const int* in_sizes, int n_in,
                              void* d_out, int out_size, void* d_ws, size_t ws_size,
                              hipStream_t stream) {
  const float* x  = (const float*)d_in[0];
  const float* Wq = (const float*)d_in[1];
  const float* Wk = (const float*)d_in[2];
  const float* Wv = (const float*)d_in[3];
  const float* Wo = (const float*)d_in[4];
  const int*  pos = (const int*)d_in[5];
  float* out = (float*)d_out;

  char* p = (char*)d_ws;
  __bf16* xb  = (__bf16*)p; p += (size_t)BS * D_MODEL * 2;        //  8 MB
  __bf16* wqb = (__bf16*)p; p += (size_t)D_MODEL * D_MODEL * 2;   //  2 MB
  __bf16* wkb = (__bf16*)p; p += (size_t)D_MODEL * D_MODEL * 2;   //  2 MB
  __bf16* wvb = (__bf16*)p; p += (size_t)D_MODEL * D_MODEL * 2;   //  2 MB
  __bf16* wob = (__bf16*)p; p += (size_t)D_MODEL * D_MODEL * 2;   //  2 MB
  float*  tmp = (float*)p;  p += (size_t)BS * D_MODEL * 4;        // 16 MB
  __bf16* qh  = (__bf16*)p; p += (size_t)BH * SEQ * HEAD_DIM * 2; //  8 MB
  __bf16* kh  = (__bf16*)p; p += (size_t)BH * SEQ * HEAD_DIM * 2; //  8 MB
  __bf16* vt  = (__bf16*)p; p += (size_t)BH * HEAD_DIM * SEQ * 2; //  8 MB
  __bf16* ob  = (__bf16*)p; p += (size_t)BS * D_MODEL * 2;        //  8 MB

  const int nX = BS * D_MODEL;          // 4,194,304
  const int nW = D_MODEL * D_MODEL;     // 1,048,576

  cvt_bf16_kernel<<<2048, 256, 0, stream>>>(x,  xb,  nX);
  cvt_bf16_kernel<<<1024, 256, 0, stream>>>(Wq, wqb, nW);
  cvt_bf16_kernel<<<1024, 256, 0, stream>>>(Wk, wkb, nW);
  cvt_bf16_kernel<<<1024, 256, 0, stream>>>(Wv, wvb, nW);
  cvt_bf16_kernel<<<1024, 256, 0, stream>>>(Wo, wob, nW);

  // (M/32)*(N/64) = 128*16 = 2048 waves -> 256 blocks of 8 waves
  const int gemmBlocks  = ((BS / 32) * (D_MODEL / 64)) / 8;
  const int ropeBlocks  = (BATCH * SEQ * NUM_HEADS * (HEAD_DIM / 2) + 255) / 256;
  const int elemBlocks  = (nX + 255) / 256;

  gemm_bf16_kernel<<<gemmBlocks, 256, 0, stream>>>(xb, wqb, tmp, BS, D_MODEL, D_MODEL);
  rope_pack_kernel<<<ropeBlocks, 256, 0, stream>>>(tmp, pos, qh);

  gemm_bf16_kernel<<<gemmBlocks, 256, 0, stream>>>(xb, wkb, tmp, BS, D_MODEL, D_MODEL);
  rope_pack_kernel<<<ropeBlocks, 256, 0, stream>>>(tmp, pos, kh);

  gemm_bf16_kernel<<<gemmBlocks, 256, 0, stream>>>(xb, wvb, tmp, BS, D_MODEL, D_MODEL);
  vpack_kernel<<<elemBlocks, 256, 0, stream>>>(tmp, vt);

  // 32 (b,h) * 128 q-tiles = 4096 waves -> 512 blocks; O^T -> tmp (reused)
  attn_kernel<<<512, 256, 0, stream>>>(qh, kh, vt, tmp);

  opack_kernel<<<elemBlocks, 256, 0, stream>>>(tmp, ob);
  gemm_bf16_kernel<<<gemmBlocks, 256, 0, stream>>>(ob, wob, out, BS, D_MODEL, D_MODEL);
}
